// Seg_SVD_LlamaAttention_8907762172503
// MI455X (gfx1250) — matrix operand order repaired and checked by reading, not compiled, not run on hardware
//
#include <hip/hip_runtime.h>

// ---------------------------------------------------------------- constants
#define S_    1024
#define H_    4096
#define NH_   32
#define HD_   128
#define RANK_ 1024
#define NCOL_ 512
#define NSVD_ (H_ - NCOL_)   // 3584

typedef __bf16 bf16;
typedef __attribute__((ext_vector_type(16))) __bf16 bf16x16;
typedef __attribute__((ext_vector_type(8)))  __bf16 bf16x8;
typedef __attribute__((ext_vector_type(8)))  float  f32x8;

__device__ __forceinline__ f32x8 zero8() {
  f32x8 z;
#pragma unroll
  for (int i = 0; i < 8; ++i) z[i] = 0.0f;
  return z;
}

// ---------------------------------------------------------------- fragments
// A fragment (16x32 bf16, M in lanes 0..15 == lanes 16..31):
//   lanes 0-15 : K = k0+0..7 (elems 0-7), k0+16..23 (elems 8-15)
//   lanes 16-31: K = k0+8..15,            k0+24..31
// rowp points at &A[m][0] with m = lane&15.
__device__ __forceinline__ bf16x16 ld_a_frag(const bf16* rowp, int k0) {
  int lane = threadIdx.x & 31;
  int off  = k0 + ((lane & 16) ? 8 : 0);
  bf16x8 lo = *(const bf16x8*)(rowp + off);
  bf16x8 hi = *(const bf16x8*)(rowp + off + 16);
  bf16x16 r;
#pragma unroll
  for (int i = 0; i < 8; ++i) { r[i] = lo[i]; r[i + 8] = hi[i]; }
  return r;
}

// B fragment (32x16 bf16, N in lanes), same K order as the A fragment:
//   lanes 0-15 : K = k0+0..7, k0+16..23 ; lanes 16-31: K = k0+8..15, k0+24..31.
// B supplied as W[N][K] row-major (we always compute X @ W^T); rowp = &W[n][0]
// with n = lane&15. Two 16-byte loads.
__device__ __forceinline__ bf16x16 ld_b_frag(const bf16* rowp, int k0) {
  int lane = threadIdx.x & 31;
  int off  = k0 + ((lane & 16) ? 8 : 0);
  bf16x8 lo = *(const bf16x8*)(rowp + off);
  bf16x8 hi = *(const bf16x8*)(rowp + off + 16);
  bf16x16 r;
#pragma unroll
  for (int i = 0; i < 8; ++i) { r[i] = lo[i]; r[i + 8] = hi[i]; }
  return r;
}

// ---------------------------------------------------------------- GEMM
// C[M][N] = A[M][K] @ W[N][K]^T  (+ Cin if addC).  One wave per 32x64 tile:
// 2 A-fragments + 4 B-fragments -> 8 WMMAs per 32-wide K chunk (~21 FLOP per
// byte loaded from L2; the whole bf16 working set is L2-resident).
// grid = (N/64, M/32), block = 32 (one wave32, EXEC all ones).
__global__ __launch_bounds__(32) void k_gemm(const bf16* __restrict__ A,
                                             const bf16* __restrict__ W,
                                             const float* __restrict__ Cin,
                                             float* __restrict__ Cout,
                                             int M, int N, int K, int addC) {
  int lane   = threadIdx.x & 31;
  int mrow   = lane & 15;
  int rowoff = (lane & 16) ? 8 : 0;     // C-frag row offset for hi half-wave
  int n0 = blockIdx.x * 64;
  int m0 = blockIdx.y * 32;

  const bf16* arow0 = A + (size_t)(m0 + mrow) * K;
  const bf16* arow1 = A + (size_t)(m0 + 16 + mrow) * K;
  const bf16* brow[4];
#pragma unroll
  for (int t = 0; t < 4; ++t) brow[t] = W + (size_t)(n0 + t * 16 + mrow) * K;

  f32x8 acc[2][4];
  if (addC) {
#pragma unroll
    for (int h = 0; h < 2; ++h)
#pragma unroll
      for (int t = 0; t < 4; ++t)
#pragma unroll
        for (int r = 0; r < 8; ++r)
          acc[h][t][r] =
              Cin[(size_t)(m0 + h * 16 + r + rowoff) * N + n0 + t * 16 + mrow];
  } else {
#pragma unroll
    for (int h = 0; h < 2; ++h)
#pragma unroll
      for (int t = 0; t < 4; ++t) acc[h][t] = zero8();
  }

  for (int k0 = 0; k0 < K; k0 += 32) {
    bf16x16 af0 = ld_a_frag(arow0, k0);
    bf16x16 af1 = ld_a_frag(arow1, k0);
#pragma unroll
    for (int t = 0; t < 4; ++t) {
      bf16x16 bfrg = ld_b_frag(brow[t], k0);
      acc[0][t] = __builtin_amdgcn_wmma_f32_16x16x32_bf16(
          false, af0, false, bfrg, (short)0, acc[0][t], false, false);
      acc[1][t] = __builtin_amdgcn_wmma_f32_16x16x32_bf16(
          false, af1, false, bfrg, (short)0, acc[1][t], false, false);
    }
  }

#pragma unroll
  for (int h = 0; h < 2; ++h)
#pragma unroll
    for (int t = 0; t < 4; ++t)
#pragma unroll
      for (int r = 0; r < 8; ++r)
        Cout[(size_t)(m0 + h * 16 + r + rowoff) * N + n0 + t * 16 + mrow] =
            acc[h][t][r];
}

// ---------------------------------------------------------------- elementwise
__global__ void k_cvt(const float* __restrict__ src, bf16* __restrict__ dst, int n) {
  int t = blockIdx.x * 256 + threadIdx.x;
  if (t < n) dst[t] = (bf16)src[t];
}

// dst[s][c] = (bf16)x[s][idx[c]]   (x row stride H_)
__global__ void k_gather(const float* __restrict__ x, const int* __restrict__ idx,
                         int ncols, bf16* __restrict__ dst) {
  int t = blockIdx.x * 256 + threadIdx.x;
  if (t >= S_ * ncols) return;
  int c = t % ncols, s = t / ncols;
  dst[t] = (bf16)x[(size_t)s * H_ + idx[c]];
}

// ---------------------------------------------------------------- RoPE + head split
// qp/kp/vp : [S][NH*HD] fp32.  Outputs: qh,kh [h][s][d] bf16 ; vt [h][d][s] bf16.
__global__ void k_rope_prep(const float* __restrict__ qp, const float* __restrict__ kp,
                            const float* __restrict__ vp, const int* __restrict__ posids,
                            bf16* __restrict__ qh, bf16* __restrict__ kh,
                            bf16* __restrict__ vt) {
  int idx = blockIdx.x * 256 + threadIdx.x;
  if (idx >= S_ * H_) return;
  int d = idx & (HD_ - 1);
  int h = (idx >> 7) & (NH_ - 1);
  int s = idx >> 12;
  float pos = (float)posids[s];
  int i = d & 63;                                   // emb = concat(ang, ang)
  float invf = __powf(10000.0f, -(float)i * (1.0f / 64.0f));
  float ang = pos * invf;
  float cs = __cosf(ang), sn = __sinf(ang);
  float q = qp[idx], k = kp[idx];
  float qr, kr;                                     // rotate_half partner
  if (d < 64) { qr = -qp[idx + 64]; kr = -kp[idx + 64]; }
  else        { qr =  qp[idx - 64]; kr =  kp[idx - 64]; }
  size_t ho = ((size_t)h * S_ + s) * HD_ + d;
  qh[ho] = (bf16)(q * cs + qr * sn);
  kh[ho] = (bf16)(k * cs + kr * sn);
  vt[((size_t)h * HD_ + d) * S_ + s] = (bf16)vp[idx];
}

// ---------------------------------------------------------------- attention
// One wave per (head, 16 query rows). 16x1024 fp32 score strip in LDS (64KB).
__global__ __launch_bounds__(32) void k_attn(const bf16* __restrict__ qh,
                                             const bf16* __restrict__ kh,
                                             const bf16* __restrict__ vt,
                                             const float* __restrict__ mask,
                                             float* __restrict__ ao) {
  __shared__ float strip[16 * S_];                  // 64 KB
  int head = blockIdx.y;
  int m0   = blockIdx.x * 16;
  int lane = threadIdx.x & 31;
  int mrow = lane & 15;
  int hioff = (lane & 16) ? 8 : 0;
  const float scale = 0.08838834764831845f;         // 1/sqrt(128)

  // Q fragments for the 16 rows (HD=128 -> 4 K-chunks of 32)
  const bf16* qrow = qh + ((size_t)head * S_ + (m0 + mrow)) * HD_;
  bf16x16 qa[4];
#pragma unroll
  for (int c = 0; c < 4; ++c) qa[c] = ld_a_frag(qrow, c * 32);

  // Phase 1: scores -> LDS (scaled + mask)
  const bf16* khead = kh + (size_t)head * S_ * HD_;
  for (int j = 0; j < S_ / 16; ++j) {
    const bf16* krow = khead + (size_t)(j * 16 + mrow) * HD_;
    if (j + 1 < S_ / 16)                     // gfx1250 global_prefetch_b8
      __builtin_prefetch(krow + 16 * HD_, 0, 1);
    f32x8 acc = zero8();
#pragma unroll
    for (int c = 0; c < 4; ++c) {
      bf16x16 bfrg = ld_b_frag(krow, c * 32);
      acc = __builtin_amdgcn_wmma_f32_16x16x32_bf16(
          false, qa[c], false, bfrg, (short)0, acc, false, false);
    }
    int col = j * 16 + mrow;
#pragma unroll
    for (int r = 0; r < 8; ++r) {
      int row = r + hioff;
      strip[row * S_ + col] =
          acc[r] * scale + mask[(size_t)(m0 + row) * S_ + col];
    }
  }
  __syncthreads();

  // Phase 2: row softmax (fp32, wave-cooperative)
  for (int r = 0; r < 16; ++r) {
    float* row = strip + r * S_;
    float m = -3.0e38f;
    for (int c = lane; c < S_; c += 32) m = fmaxf(m, row[c]);
#pragma unroll
    for (int off = 16; off; off >>= 1) m = fmaxf(m, __shfl_xor(m, off, 32));
    float l = 0.0f;
    for (int c = lane; c < S_; c += 32) { float e = __expf(row[c] - m); row[c] = e; l += e; }
#pragma unroll
    for (int off = 16; off; off >>= 1) l += __shfl_xor(l, off, 32);
    float inv = 1.0f / l;
    for (int c = lane; c < S_; c += 32) row[c] *= inv;
  }
  __syncthreads();

  // Phase 3: O = P @ V  (vt is [h][d][s] so B-fragments are contiguous in s)
  f32x8 oacc[8];
#pragma unroll
  for (int t = 0; t < 8; ++t) oacc[t] = zero8();
  const bf16* vhead = vt + (size_t)head * HD_ * S_;
  for (int kc = 0; kc < S_ / 32; ++kc) {
    int kb = kc * 32;
    // a-frag from LDS fp32 probabilities
    bf16x16 pf;
    const float* prow = strip + mrow * S_ + kb + hioff;
#pragma unroll
    for (int i = 0; i < 8; ++i) {
      pf[i]     = (bf16)prow[i];
      pf[i + 8] = (bf16)prow[i + 16];
    }
#pragma unroll
    for (int t = 0; t < 8; ++t) {
      const bf16* vrow = vhead + (size_t)(t * 16 + mrow) * S_;
      bf16x16 bfrg = ld_b_frag(vrow, kb);
      oacc[t] = __builtin_amdgcn_wmma_f32_16x16x32_bf16(
          false, pf, false, bfrg, (short)0, oacc[t], false, false);
    }
  }

  // Phase 4: store ao[s][head*HD + n] fp32
#pragma unroll
  for (int t = 0; t < 8; ++t)
#pragma unroll
    for (int r = 0; r < 8; ++r)
      ao[(size_t)(m0 + r + hioff) * H_ + head * HD_ + t * 16 + mrow] = oacc[t][r];
}

// ---------------------------------------------------------------- launch
static inline size_t alignUp256(size_t x) { return (x + 255) & ~(size_t)255; }

extern "C" void kernel_launch(void* const* d_in, const int* in_sizes, int n_in,
                              void* d_out, int out_size, void* d_ws, size_t ws_size,
                              hipStream_t stream) {
  const float* hidden = (const float*)d_in[0];
  const float* amask  = (const float*)d_in[1];
  const int*   posids = (const int*)d_in[2];
  const int*   colidx = (const int*)d_in[3];
  const int*   svdidx = (const int*)d_in[4];
  const float* wv[4] = {(const float*)d_in[5], (const float*)d_in[8],
                        (const float*)d_in[11], (const float*)d_in[14]};
  const float* wu[4] = {(const float*)d_in[6], (const float*)d_in[9],
                        (const float*)d_in[12], (const float*)d_in[15]};
  const float* wc[4] = {(const float*)d_in[7], (const float*)d_in[10],
                        (const float*)d_in[13], (const float*)d_in[16]};
  float* out = (float*)d_out;

  // Workspace carve (sequential stream ordering makes buffer reuse safe).
  unsigned char* p = (unsigned char*)d_ws;
  auto carve = [&](size_t bytes) -> void* { void* r = (void*)p; p += alignUp256(bytes); return r; };
  bf16*  xsvd  = (bf16*)carve((size_t)S_ * NSVD_ * 2);   // also reused for attn-out gather
  bf16*  xcol  = (bf16*)carve((size_t)S_ * NCOL_ * 2);
  bf16*  wvbf  = (bf16*)carve((size_t)RANK_ * NSVD_ * 2);
  bf16*  wubf  = (bf16*)carve((size_t)H_ * RANK_ * 2);
  bf16*  wcbf  = (bf16*)carve((size_t)H_ * NCOL_ * 2);
  float* rbuf  = (float*)carve((size_t)S_ * RANK_ * 4);
  bf16*  rbf   = (bf16*)carve((size_t)S_ * RANK_ * 2);
  float* proj[3];
  for (int i = 0; i < 3; ++i) proj[i] = (float*)carve((size_t)S_ * H_ * 4);
  bf16*  qhb   = (bf16*)carve((size_t)S_ * H_ * 2);
  bf16*  khb   = (bf16*)carve((size_t)S_ * H_ * 2);
  bf16*  vtb   = (bf16*)carve((size_t)S_ * H_ * 2);

  const int TPB = 256;
  auto blocks = [](long n) { return (unsigned)((n + 255) / 256); };
  dim3 gN64(H_ / 64, S_ / 32);      // N = 4096 tiles, 32-row M blocks
  dim3 gR64(RANK_ / 64, S_ / 32);   // N = 1024 tiles

  // Gather + bf16-convert the two activation slices of hidden_states.
  k_gather<<<blocks((long)S_ * NSVD_), TPB, 0, stream>>>(hidden, svdidx, NSVD_, xsvd);
  k_gather<<<blocks((long)S_ * NCOL_), TPB, 0, stream>>>(hidden, colidx, NCOL_, xcol);

  // q, k, v projections.
  for (int pi = 0; pi < 3; ++pi) {
    k_cvt<<<blocks((long)RANK_ * NSVD_), TPB, 0, stream>>>(wv[pi], wvbf, RANK_ * NSVD_);
    k_gemm<<<gR64, 32, 0, stream>>>(xsvd, wvbf, nullptr, rbuf, S_, RANK_, NSVD_, 0);
    k_cvt<<<blocks((long)S_ * RANK_), TPB, 0, stream>>>(rbuf, rbf, S_ * RANK_);
    k_cvt<<<blocks((long)H_ * NCOL_), TPB, 0, stream>>>(wc[pi], wcbf, H_ * NCOL_);
    k_gemm<<<gN64, 32, 0, stream>>>(xcol, wcbf, nullptr, proj[pi], S_, H_, NCOL_, 0);
    k_cvt<<<blocks((long)H_ * RANK_), TPB, 0, stream>>>(wu[pi], wubf, H_ * RANK_);
    k_gemm<<<gN64, 32, 0, stream>>>(rbf, wubf, proj[pi], proj[pi], S_, H_, RANK_, 1);
  }

  // RoPE + head split (+ V transpose).
  k_rope_prep<<<blocks((long)S_ * H_), TPB, 0, stream>>>(
      proj[0], proj[1], proj[2], posids, qhb, khb, vtb);

  // Attention (proj[0] is free now; reuse as attention output).
  float* ao = proj[0];
  k_attn<<<dim3(S_ / 16, NH_), 32, 0, stream>>>(qhb, khb, vtb, amask, ao);

  // o projection into d_out.
  k_gather<<<blocks((long)S_ * NSVD_), TPB, 0, stream>>>(ao, svdidx, NSVD_, xsvd);
  k_gather<<<blocks((long)S_ * NCOL_), TPB, 0, stream>>>(ao, colidx, NCOL_, xcol);
  k_cvt<<<blocks((long)RANK_ * NSVD_), TPB, 0, stream>>>(wv[3], wvbf, RANK_ * NSVD_);
  k_gemm<<<gR64, 32, 0, stream>>>(xsvd, wvbf, nullptr, rbuf, S_, RANK_, NSVD_, 0);
  k_cvt<<<blocks((long)S_ * RANK_), TPB, 0, stream>>>(rbuf, rbf, S_ * RANK_);
  k_cvt<<<blocks((long)H_ * NCOL_), TPB, 0, stream>>>(wc[3], wcbf, H_ * NCOL_);
  k_gemm<<<gN64, 32, 0, stream>>>(xcol, wcbf, nullptr, out, S_, H_, NCOL_, 0);
  k_cvt<<<blocks((long)H_ * RANK_), TPB, 0, stream>>>(wu[3], wubf, H_ * RANK_);
  k_gemm<<<gN64, 32, 0, stream>>>(rbf, wubf, out, out, S_, H_, RANK_, 1);

  (void)in_sizes; (void)n_in; (void)out_size; (void)ws_size;
}
